// CrossAttentionFusion_26190710571282
// MI455X (gfx1250) — compile-verified, matrix operand order checked by reading, not run on hardware
//
#include <hip/hip_runtime.h>
#include <math.h>

// ---------------------------------------------------------------------------
// CDNA5 (gfx1250) cross-attention fusion.
// B=64, D=512, H=4 (hd=128), MAX_MOL=128, MAX_TGT=1024.
// All matmuls via v_wmma_f32_16x16x32_f16; each wave owns a 32x32 output tile
// (2 A-frags x 2 B-frags -> 4 WMMAs per K-step) for 2x operand reuse.
// ---------------------------------------------------------------------------

typedef __attribute__((ext_vector_type(16))) _Float16 v16h;
typedef __attribute__((ext_vector_type(8)))  _Float16 v8h;
typedef __attribute__((ext_vector_type(4)))  _Float16 v4h;
typedef __attribute__((ext_vector_type(8)))  float    v8f;

#define BSZ      64
#define DIM      512
#define NHEAD    4
#define HDIM     128
#define NMOL     128
#define NTGT     1024

// ---------------------------------------------------------------------------
// Generic batched WMMA GEMM: out[m,n] = act( scale * sum_k A[m,k]*Bw[n,k]
//                                            + bias[n] + resid[m,n] )
// A, Bw fp16 row-major. One wave computes a 32x32 tile; K-loop step 32.
// blockIdx.y = composite batch (b*nH + h); per-(b,h) pointer strides.
// M,N multiples of 32; Kd multiple of 32 (true for all uses here).
// ---------------------------------------------------------------------------
__global__ __launch_bounds__(256)
void wmma_gemm(const _Float16* __restrict__ A, const _Float16* __restrict__ Bw,
               const float* __restrict__ bias, const float* __restrict__ resid,
               float* __restrict__ C, _Float16* __restrict__ C16,
               int M, int N, int Kd, int lda, int ldb, int ldc,
               long sAb, long sAh, long sBb, long sBh, long sCb, long sCh,
               int nH, float scale, int act)
{
    const int batch = blockIdx.y;
    const int b = batch / nH;
    const int h = batch % nH;
    A  += (long)b * sAb + (long)h * sAh;
    Bw += (long)b * sBb + (long)h * sBh;
    const long coff = (long)b * sCb + (long)h * sCh;

    const int tilesN = N >> 5;                  // 32-wide tiles
    const int tiles  = (M >> 5) * tilesN;
    const int wave   = blockIdx.x * (blockDim.x >> 5) + (threadIdx.x >> 5);
    if (wave >= tiles) return;                  // uniform per wave: EXEC stays full
    const int tm = wave / tilesN;
    const int tn = wave % tilesN;

    const int lane = threadIdx.x & 31;
    const int ll   = lane & 15;                 // row within 16-row group
    const int hi   = lane >> 4;                 // K half-select per ISA layout

    const _Float16* ar0 = A  + (long)(tm * 32 + ll) * lda;
    const _Float16* ar1 = ar0 + (long)16 * lda;
    const _Float16* br0 = Bw + (long)(tn * 32 + ll) * ldb;
    const _Float16* br1 = br0 + (long)16 * ldb;

    v8f acc00 = {}, acc01 = {}, acc10 = {}, acc11 = {};
    for (int k = 0; k < Kd; k += 32) {
        union { v16h v; v8h h2[2]; } a0, a1, b0, b1;
        // 16-bit operand layout: lane<16 -> K = {k..k+7, k+16..k+23}; lane>=16 -> +8
        a0.h2[0] = *(const v8h*)(ar0 + k + hi * 8);
        a0.h2[1] = *(const v8h*)(ar0 + k + 16 + hi * 8);
        a1.h2[0] = *(const v8h*)(ar1 + k + hi * 8);
        a1.h2[1] = *(const v8h*)(ar1 + k + 16 + hi * 8);
        b0.h2[0] = *(const v8h*)(br0 + k + hi * 8);
        b0.h2[1] = *(const v8h*)(br0 + k + 16 + hi * 8);
        b1.h2[0] = *(const v8h*)(br1 + k + hi * 8);
        b1.h2[1] = *(const v8h*)(br1 + k + 16 + hi * 8);
        if (k + 32 < Kd) {                      // lowers to global_prefetch_b8
            __builtin_prefetch(ar0 + k + 32, 0, 1);
            __builtin_prefetch(ar1 + k + 32, 0, 1);
            __builtin_prefetch(br0 + k + 32, 0, 1);
            __builtin_prefetch(br1 + k + 32, 0, 1);
        }
        acc00 = __builtin_amdgcn_wmma_f32_16x16x32_f16(false, a0.v, false, b0.v, (short)0, acc00, false, false);
        acc01 = __builtin_amdgcn_wmma_f32_16x16x32_f16(false, a0.v, false, b1.v, (short)0, acc01, false, false);
        acc10 = __builtin_amdgcn_wmma_f32_16x16x32_f16(false, a1.v, false, b0.v, (short)0, acc10, false, false);
        acc11 = __builtin_amdgcn_wmma_f32_16x16x32_f16(false, a1.v, false, b1.v, (short)0, acc11, false, false);
    }

    // C/D layout: lane holds N = ll within tile; VGPR r holds M = r + hi*8
    const int col0 = tn * 32 + ll;
    const int col1 = col0 + 16;
    const float bv0 = bias ? bias[col0] : 0.f;
    const float bv1 = bias ? bias[col1] : 0.f;
    const v8f* accs[4] = { &acc00, &acc01, &acc10, &acc11 };
#pragma unroll
    for (int t = 0; t < 4; ++t) {
        const int row_base = tm * 32 + (t >> 1) * 16;
        const int col = (t & 1) ? col1 : col0;
        const float bv = (t & 1) ? bv1 : bv0;
        const v8f a = *accs[t];
#pragma unroll
        for (int r = 0; r < 8; ++r) {
            const int row = row_base + r + hi * 8;
            float v = a[r] * scale + bv;
            if (resid) v += resid[(long)row * ldc + col];
            if (act)   v = 0.5f * v * (1.f + erff(v * 0.70710678118f));  // exact GELU
            const long o = coff + (long)row * ldc + col;
            if (C)   C[o]   = v;
            if (C16) C16[o] = (_Float16)v;
        }
    }
}

// ---------------------------------------------------------------------------
// Softmax over 1024 keys for all 4 heads of one (batch, query-row); writes
// fp16 probs (WMMA A-operand for probs@V) and the head-mean directly to d_out.
// One block (256 threads) per (b, n); 4 columns per thread.
// ---------------------------------------------------------------------------
__global__ __launch_bounds__(256)
void softmax_wmean(const float* __restrict__ scores, _Float16* __restrict__ probs,
                   float* __restrict__ wmean)
{
    __shared__ float red[256];
    const int tid = threadIdx.x;
    const int b = blockIdx.x >> 7;
    const int n = blockIdx.x & 127;
    float wacc0 = 0.f, wacc1 = 0.f, wacc2 = 0.f, wacc3 = 0.f;

    for (int h = 0; h < NHEAD; ++h) {
        const long base = (((long)(b * NHEAD + h)) * NMOL + n) * NTGT + tid * 4;
        const float4 s = *(const float4*)(scores + base);
        float m = fmaxf(fmaxf(s.x, s.y), fmaxf(s.z, s.w));
        red[tid] = m; __syncthreads();
        for (int st = 128; st > 0; st >>= 1) {
            if (tid < st) red[tid] = fmaxf(red[tid], red[tid + st]);
            __syncthreads();
        }
        m = red[0]; __syncthreads();

        const float e0 = __expf(s.x - m), e1 = __expf(s.y - m);
        const float e2 = __expf(s.z - m), e3 = __expf(s.w - m);
        red[tid] = e0 + e1 + e2 + e3; __syncthreads();
        for (int st = 128; st > 0; st >>= 1) {
            if (tid < st) red[tid] += red[tid + st];
            __syncthreads();
        }
        const float inv = 1.f / red[0]; __syncthreads();

        const float p0 = e0 * inv, p1 = e1 * inv, p2 = e2 * inv, p3 = e3 * inv;
        v4h pk; pk[0] = (_Float16)p0; pk[1] = (_Float16)p1;
                pk[2] = (_Float16)p2; pk[3] = (_Float16)p3;
        *(v4h*)(probs + base) = pk;
        wacc0 += p0 * 0.25f; wacc1 += p1 * 0.25f;
        wacc2 += p2 * 0.25f; wacc3 += p3 * 0.25f;
    }
    float4 wo; wo.x = wacc0; wo.y = wacc1; wo.z = wacc2; wo.w = wacc3;
    *(float4*)(wmean + (long)blockIdx.x * NTGT + tid * 4) = wo;
}

// ---------------------------------------------------------------------------
// LayerNorm over D=512; one block per row, 2 columns per thread.
// ---------------------------------------------------------------------------
__global__ __launch_bounds__(256)
void layernorm_rows(const float* __restrict__ x, const float* __restrict__ g,
                    const float* __restrict__ be, float* __restrict__ out)
{
    __shared__ float red[256];
    const int tid = threadIdx.x;
    const long row = blockIdx.x;
    const float* xr = x + row * DIM;
    const float a = xr[tid], c = xr[tid + 256];

    red[tid] = a + c; __syncthreads();
    for (int st = 128; st > 0; st >>= 1) {
        if (tid < st) red[tid] += red[tid + st];
        __syncthreads();
    }
    const float mu = red[0] * (1.f / DIM); __syncthreads();

    const float da = a - mu, dc = c - mu;
    red[tid] = da * da + dc * dc; __syncthreads();
    for (int st = 128; st > 0; st >>= 1) {
        if (tid < st) red[tid] += red[tid + st];
        __syncthreads();
    }
    const float rs = rsqrtf(red[0] * (1.f / DIM) + 1e-5f);

    out[row * DIM + tid]       = da * rs * g[tid]       + be[tid];
    out[row * DIM + tid + 256] = dc * rs * g[tid + 256] + be[tid + 256];
}

// --------------------------- small utility kernels -------------------------
__global__ void zero_u32(unsigned* __restrict__ p, long n) {
    const long i = (long)blockIdx.x * blockDim.x + threadIdx.x;
    if (i < n) p[i] = 0u;
}

__global__ void f32_to_f16(const float* __restrict__ s, _Float16* __restrict__ d, long n) {
    const long i = (long)blockIdx.x * blockDim.x + threadIdx.x;
    if (i < n) d[i] = (_Float16)s[i];
}

__global__ void scatter_mol(const float* __restrict__ src, const int* __restrict__ batch,
                            const int* __restrict__ pos, float* __restrict__ padF,
                            _Float16* __restrict__ padH, long total)
{
    const long i = (long)blockIdx.x * blockDim.x + threadIdx.x;
    if (i >= total) return;
    const long r = i >> 9;
    const int  c = (int)(i & 511);
    const float v = src[i];
    const long o = ((long)batch[r] * NMOL + pos[r]) * DIM + c;
    padF[o] = v;
    padH[o] = (_Float16)v;
}

__global__ void scatter_tgt(const float* __restrict__ src, const int* __restrict__ batch,
                            const int* __restrict__ pos, _Float16* __restrict__ padH, long total)
{
    const long i = (long)blockIdx.x * blockDim.x + threadIdx.x;
    if (i >= total) return;
    const long r = i >> 9;
    const int  c = (int)(i & 511);
    const long o = ((long)batch[r] * NTGT + pos[r]) * DIM + c;
    padH[o] = (_Float16)src[i];
}

// V [b][1024][512] -> Vt [b][512][1024] so probs@V uses row-contiguous B operand
__global__ void transpose_v(const _Float16* __restrict__ V, _Float16* __restrict__ Vt, long total)
{
    const long i = (long)blockIdx.x * blockDim.x + threadIdx.x;
    if (i >= total) return;
    const long b   = i / ((long)NTGT * DIM);
    const long rem = i - b * (long)NTGT * DIM;
    const int  m   = (int)(rem >> 9);
    const int  d   = (int)(rem & 511);
    Vt[b * (long)DIM * NTGT + (long)d * NTGT + m] = V[i];
}

// pooled[b][d] = mean over 128 rows of h_tilde
__global__ void pool_mean(const float* __restrict__ ht, float* __restrict__ pooled, long total)
{
    const long i = (long)blockIdx.x * blockDim.x + threadIdx.x;
    if (i >= total) return;
    const int b = (int)(i >> 9);
    const int d = (int)(i & 511);
    const float* h = ht + (long)b * NMOL * DIM + d;
    float s = 0.f;
    for (int n = 0; n < NMOL; ++n) s += h[(long)n * DIM];
    pooled[i] = s * (1.f / NMOL);
}

// fused_f16[b][0:512]=mol_graph, [512:1024]=target_graph, [1024:1536]=pooled
__global__ void build_fused(const float* __restrict__ molg, const float* __restrict__ tgtg,
                            const float* __restrict__ pooled, _Float16* __restrict__ out, long total)
{
    const long i = (long)blockIdx.x * blockDim.x + threadIdx.x;
    if (i >= total) return;
    const int b = (int)(i / (3 * DIM));
    const int c = (int)(i % (3 * DIM));
    float v;
    if (c < DIM)            v = molg[(long)b * DIM + c];
    else if (c < 2 * DIM)   v = tgtg[(long)b * DIM + (c - DIM)];
    else                    v = pooled[(long)b * DIM + (c - 2 * DIM)];
    out[i] = (_Float16)v;
}

// ---------------------------------------------------------------------------
extern "C" void kernel_launch(void* const* d_in, const int* in_sizes, int n_in,
                              void* d_out, int out_size, void* d_ws, size_t ws_size,
                              hipStream_t stream)
{
    (void)n_in; (void)out_size; (void)ws_size;

    const float* mol_node = (const float*)d_in[0];
    const float* tgt_node = (const float*)d_in[1];
    const float* molg     = (const float*)d_in[2];
    const float* tgtg     = (const float*)d_in[3];
    const float* Wq = (const float*)d_in[4];  const float* bq = (const float*)d_in[5];
    const float* Wk = (const float*)d_in[6];  const float* bk = (const float*)d_in[7];
    const float* Wv = (const float*)d_in[8];  const float* bv = (const float*)d_in[9];
    const float* Wo = (const float*)d_in[10]; const float* bo = (const float*)d_in[11];
    const float* ln_g = (const float*)d_in[12];
    const float* ln_b = (const float*)d_in[13];
    const float* Wf1 = (const float*)d_in[14]; const float* bf1 = (const float*)d_in[15];
    const float* Wf2 = (const float*)d_in[16]; const float* bf2 = (const float*)d_in[17];
    const int* mol_batch = (const int*)d_in[18];
    const int* mol_pos   = (const int*)d_in[19];
    const int* tgt_batch = (const int*)d_in[20];
    const int* tgt_pos   = (const int*)d_in[21];

    const long nMolElem = in_sizes[0];      // n_mol_nodes * 512
    const long nTgtElem = in_sizes[1];      // n_tgt_nodes * 512

    // ----- workspace carve (256B aligned) -----
    char* wp = (char*)d_ws;
    auto carve = [&](size_t bytes) -> void* {
        void* p = (void*)wp;
        wp += (bytes + 255) & ~(size_t)255;
        return p;
    };
    float*    molPadF = (float*)   carve((size_t)BSZ * NMOL * DIM * 4);
    _Float16* molPadH = (_Float16*)carve((size_t)BSZ * NMOL * DIM * 2);
    _Float16* tgtPadH = (_Float16*)carve((size_t)BSZ * NTGT * DIM * 2);
    _Float16* Qh      = (_Float16*)carve((size_t)BSZ * NMOL * DIM * 2);
    _Float16* Kh      = (_Float16*)carve((size_t)BSZ * NTGT * DIM * 2);
    _Float16* Vh      = (_Float16*)carve((size_t)BSZ * NTGT * DIM * 2);
    _Float16* Vt      = (_Float16*)carve((size_t)BSZ * DIM * NTGT * 2);
    float*    scoresF = (float*)   carve((size_t)BSZ * NHEAD * NMOL * NTGT * 4);
    _Float16* probsH  = (_Float16*)carve((size_t)BSZ * NHEAD * NMOL * NTGT * 2);
    _Float16* attH    = (_Float16*)carve((size_t)BSZ * NMOL * DIM * 2);
    float*    xbuf    = (float*)   carve((size_t)BSZ * NMOL * DIM * 4);
    float*    pooled  = (float*)   carve((size_t)BSZ * DIM * 4);
    _Float16* fusedH  = (_Float16*)carve((size_t)BSZ * 3 * DIM * 2);
    _Float16* gH      = (_Float16*)carve((size_t)BSZ * DIM * 2);
    _Float16* WqH = (_Float16*)carve((size_t)DIM * DIM * 2);
    _Float16* WkH = (_Float16*)carve((size_t)DIM * DIM * 2);
    _Float16* WvH = (_Float16*)carve((size_t)DIM * DIM * 2);
    _Float16* WoH = (_Float16*)carve((size_t)DIM * DIM * 2);
    _Float16* Wf1H = (_Float16*)carve((size_t)DIM * 3 * DIM * 2);
    _Float16* Wf2H = (_Float16*)carve((size_t)DIM * DIM * 2);

    float* outFused = (float*)d_out;                                     // (64,512)
    float* outHt    = outFused + (size_t)BSZ * DIM;                      // (64,128,512)
    float* outWmean = outHt + (size_t)BSZ * NMOL * DIM;                  // (64,128,1024)

    auto blocks = [](long n, int t) { return (unsigned)((n + t - 1) / t); };

    // ----- weight fp32 -> fp16 -----
    f32_to_f16<<<blocks(DIM * DIM, 256), 256, 0, stream>>>(Wq, WqH, DIM * DIM);
    f32_to_f16<<<blocks(DIM * DIM, 256), 256, 0, stream>>>(Wk, WkH, DIM * DIM);
    f32_to_f16<<<blocks(DIM * DIM, 256), 256, 0, stream>>>(Wv, WvH, DIM * DIM);
    f32_to_f16<<<blocks(DIM * DIM, 256), 256, 0, stream>>>(Wo, WoH, DIM * DIM);
    f32_to_f16<<<blocks((long)DIM * 3 * DIM, 256), 256, 0, stream>>>(Wf1, Wf1H, (long)DIM * 3 * DIM);
    f32_to_f16<<<blocks(DIM * DIM, 256), 256, 0, stream>>>(Wf2, Wf2H, DIM * DIM);

    // ----- zero pads, scatter ragged -> padded -----
    zero_u32<<<blocks((long)BSZ * NMOL * DIM, 256), 256, 0, stream>>>((unsigned*)molPadF, (long)BSZ * NMOL * DIM);
    zero_u32<<<blocks((long)BSZ * NMOL * DIM / 2, 256), 256, 0, stream>>>((unsigned*)molPadH, (long)BSZ * NMOL * DIM / 2);
    zero_u32<<<blocks((long)BSZ * NTGT * DIM / 2, 256), 256, 0, stream>>>((unsigned*)tgtPadH, (long)BSZ * NTGT * DIM / 2);
    scatter_mol<<<blocks(nMolElem, 256), 256, 0, stream>>>(mol_node, mol_batch, mol_pos, molPadF, molPadH, nMolElem);
    scatter_tgt<<<blocks(nTgtElem, 256), 256, 0, stream>>>(tgt_node, tgt_batch, tgt_pos, tgtPadH, nTgtElem);

    auto gemm = [&](const _Float16* A, const _Float16* Bw, const float* bias,
                    const float* resid, float* C, _Float16* C16,
                    int M, int N, int Kd, int lda, int ldb, int ldc,
                    long sAb, long sAh, long sBb, long sBh, long sCb, long sCh,
                    int nH, int nBatch, float scale, int act) {
        const int tiles = (M / 32) * (N / 32);
        dim3 g((tiles + 7) / 8, nBatch);
        wmma_gemm<<<g, 256, 0, stream>>>(A, Bw, bias, resid, C, C16,
                                         M, N, Kd, lda, ldb, ldc,
                                         sAb, sAh, sBb, sBh, sCb, sCh,
                                         nH, scale, act);
    };

    // ----- projections: X @ W^T + b  (fp16 out for WMMA consumers) -----
    gemm(molPadH, WqH, bq, nullptr, nullptr, Qh,
         BSZ * NMOL, DIM, DIM, DIM, DIM, DIM, 0, 0, 0, 0, 0, 0, 1, 1, 1.f, 0);
    gemm(tgtPadH, WkH, bk, nullptr, nullptr, Kh,
         BSZ * NTGT, DIM, DIM, DIM, DIM, DIM, 0, 0, 0, 0, 0, 0, 1, 1, 1.f, 0);
    gemm(tgtPadH, WvH, bv, nullptr, nullptr, Vh,
         BSZ * NTGT, DIM, DIM, DIM, DIM, DIM, 0, 0, 0, 0, 0, 0, 1, 1, 1.f, 0);
    transpose_v<<<blocks((long)BSZ * NTGT * DIM, 256), 256, 0, stream>>>(Vh, Vt, (long)BSZ * NTGT * DIM);

    // ----- scores = Q K^T / sqrt(hd), per (b,h): (128x128)x(128x1024) -----
    gemm(Qh, Kh, nullptr, nullptr, scoresF, nullptr,
         NMOL, NTGT, HDIM, DIM, DIM, NTGT,
         (long)NMOL * DIM, HDIM,                         // Q strides: b, h
         (long)NTGT * DIM, HDIM,                         // K strides: b, h
         (long)NHEAD * NMOL * NTGT, (long)NMOL * NTGT,   // C strides: b, h
         NHEAD, BSZ * NHEAD, 0.08838834764831845f, 0);

    // ----- softmax + head-mean (w.mean(axis=1) straight to d_out) -----
    softmax_wmean<<<BSZ * NMOL, 256, 0, stream>>>(scoresF, probsH, outWmean);

    // ----- att = probs @ V, per (b,h): (128x1024)x(1024x128) -----
    gemm(probsH, Vt, nullptr, nullptr, nullptr, attH,
         NMOL, HDIM, NTGT, NTGT, NTGT, DIM,
         (long)NHEAD * NMOL * NTGT, (long)NMOL * NTGT,   // probs strides: b, h
         (long)DIM * NTGT, (long)HDIM * NTGT,            // Vt strides: b, h
         (long)NMOL * DIM, HDIM,                         // att strides: b, h
         NHEAD, BSZ * NHEAD, 1.f, 0);

    // ----- x = att @ Wo^T + bo + mol_pad (residual) -----
    gemm(attH, WoH, bo, molPadF, xbuf, nullptr,
         BSZ * NMOL, DIM, DIM, DIM, DIM, DIM, 0, 0, 0, 0, 0, 0, 1, 1, 1.f, 0);

    // ----- layernorm -> h_tilde (output), pooled mean -----
    layernorm_rows<<<BSZ * NMOL, 256, 0, stream>>>(xbuf, ln_g, ln_b, outHt);
    pool_mean<<<blocks((long)BSZ * DIM, 256), 256, 0, stream>>>(outHt, pooled, (long)BSZ * DIM);

    // ----- fused MLP: concat -> W1^T + gelu -> W2^T -----
    build_fused<<<blocks((long)BSZ * 3 * DIM, 256), 256, 0, stream>>>(molg, tgtg, pooled, fusedH, (long)BSZ * 3 * DIM);
    gemm(fusedH, Wf1H, bf1, nullptr, nullptr, gH,
         BSZ, DIM, 3 * DIM, 3 * DIM, 3 * DIM, DIM, 0, 0, 0, 0, 0, 0, 1, 1, 1.f, 1);
    gemm(gH, Wf2H, bf2, nullptr, outFused, nullptr,
         BSZ, DIM, DIM, DIM, DIM, DIM, 0, 0, 0, 0, 0, 0, 1, 1, 1.f, 0);
}
